// SingleLevelGraph_43276090474714
// MI455X (gfx1250) — matrix operand order repaired and checked by reading, not compiled, not run on hardware
//
#include <hip/hip_runtime.h>
#include <math.h>

typedef unsigned short ushort_t;
typedef __attribute__((ext_vector_type(16))) __bf16    v16bf;
typedef __attribute__((ext_vector_type(8)))  float     v8f;

struct U32x8 { uint4 a, b; };

// Async global->LDS staging: inline asm (gfx1250 VFLAT async op + ASYNCcnt wait).
// Only enabled in the device pass so the host semantic pass never sees AMDGPU asm.
#if defined(__HIP_DEVICE_COMPILE__)
#define USE_ASYNC_LDS 1
#else
#define USE_ASYNC_LDS 0
#endif

__device__ __forceinline__ ushort_t f2bfu(float f) {
    unsigned u = __builtin_bit_cast(unsigned, f);
    return (ushort_t)((u + 0x7FFFu + ((u >> 16) & 1u)) >> 16);   // RNE
}
__device__ __forceinline__ unsigned pack2(float lo, float hi) {
    return (unsigned)f2bfu(lo) | ((unsigned)f2bfu(hi) << 16);
}

// ---------------------------------------------------------------------------
// One-time weight pack: W[K,N] fp32 -> Wp[(s*N + n)*16 + kp] u32, where the
// u32 holds bf16 pair {k = 32s+2kp (lo16), k+1 (hi16)} of column n.  This is
// k-ascending per column; a lane's B fragment is u32 {4h..4h+3, 8+4h..8+4h+3}, so the
// GEMM stages B with raw 16-byte copies (no conversion in the hot loop).
// ---------------------------------------------------------------------------
__global__ void pack_weight(const float* __restrict__ W, unsigned* __restrict__ Wp,
                            int K, int N)
{
    const int idx = blockIdx.x * blockDim.x + threadIdx.x;      // over (K/2)*N
    if (idx >= (K >> 1) * N) return;
    const int kp   = idx & 15;
    const int rest = idx >> 4;
    const int n    = rest % N;
    const int s    = rest / N;
    const int k    = s * 32 + 2 * kp;
    Wp[idx] = pack2(W[(size_t)k * N + n], W[(size_t)(k + 1) * N + n]);
}

// ---------------------------------------------------------------------------
// bf16-WMMA GEMM, fp32 accumulate:  C = op(A_bf16[M,K] @ W + bias)
//   A  : bf16 row-major.   Wp : pre-packed fragment layout (above).
//   SF : store fp32 to Cf (ldOut, with REMAP row mapping)
//   SB : store bf16 row-major (ld = N) to Cb
//   REMAP: 0 direct ; 1 fine ((m>>10)*1025 + (m&1023) + 1) ; 2 global (m*1025)
// Block = 256 thr (8 waves); wave tile 16x64; block tile 128x64.
// B tile double-buffered in LDS, filled by async global->LDS copies that
// overlap the WMMAs of the previous tile.
// ---------------------------------------------------------------------------
template <int RELU, int REMAP, int SF, int SB>
__global__ void gemm_wmma(const ushort_t* __restrict__ A, const unsigned* __restrict__ Wp,
                          const float* __restrict__ bias, float* __restrict__ Cf,
                          ushort_t* __restrict__ Cb, int M, int N, int K, int ldOut)
{
    constexpr int LROW = 20;                       // u32 row stride (pad vs bank conflicts)
    __shared__ __align__(16) unsigned Bs[2][64][LROW];

    const int lane  = threadIdx.x & 31;
    const int wave  = threadIdx.x >> 5;
    const int mBase = blockIdx.y * 128 + wave * 16;
    const int nBase = blockIdx.x * 64;

    const int half = lane >> 4;                    // 0: lanes 0-15, 1: lanes 16-31
    const int l15  = lane & 15;
    const int m    = mBase + l15;
    const int aRow = (m < M) ? m : (M - 1);

    const int cpn = threadIdx.x >> 2;              // copy: column 0..63
    const int cpq = threadIdx.x & 3;               // copy: 16B quarter 0..3
    const int S   = K >> 5;                        // k-steps

    // issue copy of tile s into LDS buffer `buf` (16B per thread, 4KB per tile)
    auto issue_copy = [&](int s, int buf) {
        const unsigned* src = Wp + ((size_t)s * N + nBase + cpn) * 16 + cpq * 4;
        unsigned* dst = &Bs[buf][cpn][cpq * 4];
#if USE_ASYNC_LDS
        const unsigned           ldsa = (unsigned)(unsigned long long)dst;
        const unsigned long long ga   = (unsigned long long)src;
        asm volatile("global_load_async_to_lds_b128 %0, %1, off"
                     :: "v"(ldsa), "v"(ga) : "memory");
#else
        *(uint4*)dst = *(const uint4*)src;
#endif
    };

    v8f acc[4] = {};
    issue_copy(0, 0);

    for (int s = 0; s < S; ++s) {
        const int cur = s & 1;
#if USE_ASYNC_LDS
        asm volatile("s_wait_asynccnt 0" ::: "memory");   // own copy of tile s landed
#endif
        __syncthreads();                           // everyone's copy landed; prev reads done
        if (s + 1 < S) issue_copy(s + 1, cur ^ 1); // overlaps with the WMMAs below

        // A fragment: lane holds M=aRow, K = {kh..kh+7, kh+16..kh+23}, kh = 8*half
        const ushort_t* ap = A + (size_t)aRow * K + s * 32 + 8 * half;
        U32x8 araw;
        araw.a = *(const uint4*)(ap);
        araw.b = *(const uint4*)(ap + 16);
        const v16bf av = __builtin_bit_cast(v16bf, araw);

        // preload all 4 B fragments, then run the 4 WMMAs
        U32x8 braw[4];
#pragma unroll
        for (int t = 0; t < 4; ++t) {
            const unsigned* bp = &Bs[cur][t * 16 + l15][half * 4];
            braw[t].a = *(const uint4*)(bp);
            braw[t].b = *(const uint4*)(bp + 8);
        }
#pragma unroll
        for (int t = 0; t < 4; ++t) {
            acc[t] = __builtin_amdgcn_wmma_f32_16x16x32_bf16(
                false, av, false, __builtin_bit_cast(v16bf, braw[t]),
                (short)0, acc[t], false, false);
        }
    }

    // ---- epilogue: bias / relu / remapped fp32 and/or bf16 stores ----
#pragma unroll
    for (int t = 0; t < 4; ++t) {
        const int n  = nBase + t * 16 + l15;
        const float bv = bias ? bias[n] : 0.0f;
#pragma unroll
        for (int r = 0; r < 8; ++r) {
            const int mm = mBase + half * 8 + r;
            if (mm < M) {
                float v = acc[t][r] + bv;
                if (RELU) v = fmaxf(v, 0.0f);
                if (SF) {
                    size_t row;
                    if (REMAP == 1)      row = (size_t)(mm >> 10) * 1025 + (size_t)(mm & 1023) + 1;
                    else if (REMAP == 2) row = (size_t)mm * 1025;
                    else                 row = (size_t)mm;
                    Cf[row * (size_t)ldOut + n] = v;
                }
                if (SB) Cb[(size_t)mm * N + n] = f2bfu(v);
            }
        }
    }
}

// ---------------------------------------------------------------------------
// Row-wise LayerNorm; optional fp32 and/or bf16 outputs.
// ---------------------------------------------------------------------------
__global__ void ln_kernel(const float* __restrict__ x, const float* __restrict__ g,
                          const float* __restrict__ b, float* __restrict__ yf,
                          ushort_t* __restrict__ yb, int C)
{
    __shared__ float red[256];
    const int row = blockIdx.x;
    const float* xr = x + (size_t)row * C;
    float s = 0.f, s2 = 0.f;
    for (int c = threadIdx.x; c < C; c += 256) { float v = xr[c]; s += v; s2 += v * v; }

    red[threadIdx.x] = s; __syncthreads();
    for (int off = 128; off > 0; off >>= 1) {
        if (threadIdx.x < off) red[threadIdx.x] += red[threadIdx.x + off];
        __syncthreads();
    }
    const float mean = red[0] / C;
    __syncthreads();

    red[threadIdx.x] = s2; __syncthreads();
    for (int off = 128; off > 0; off >>= 1) {
        if (threadIdx.x < off) red[threadIdx.x] += red[threadIdx.x + off];
        __syncthreads();
    }
    const float var  = red[0] / C - mean * mean;
    const float rstd = rsqrtf(var + 1e-5f);

    for (int c = threadIdx.x; c < C; c += 256) {
        const float v = (xr[c] - mean) * rstd * g[c] + b[c];
        if (yf) yf[(size_t)row * C + c] = v;
        if (yb) yb[(size_t)row * C + c] = f2bfu(v);
    }
}

// ---------------------------------------------------------------------------
// Per-head attention dot: out[i,h] = sum_c x[i, h*64+c] * att[h*64+c]
// ---------------------------------------------------------------------------
__global__ void attn_dot(const float* __restrict__ x, const float* __restrict__ att,
                         float* __restrict__ out, int nrows)
{
    const int idx = blockIdx.x * blockDim.x + threadIdx.x;
    if (idx >= nrows * 8) return;
    const int i = idx >> 3, h = idx & 7;
    const float* xp = x + (size_t)i * 512 + h * 64;
    const float* ap = att + h * 64;
    float s = 0.f;
#pragma unroll 8
    for (int c = 0; c < 64; ++c) s += xp[c] * ap[c];
    out[idx] = s;
}

// ---------------------------------------------------------------------------
// GAT segment softmax + aggregation + bias + residual.  One block (512 thr)
// per graph b.  Star topology: edges (b*1024+e) -> b, plus self loop (source
// = fine node b); the pre-existing self-edge (row==col, only i==0) is masked.
// Emits xg_out as bf16 (A input of the `og` GEMM).
// ---------------------------------------------------------------------------
__global__ void gat_aggregate(const float* __restrict__ a_s, const float* __restrict__ a_d,
                              const float* __restrict__ xs,  const float* __restrict__ xg_ln,
                              const float* __restrict__ gat_bias, ushort_t* __restrict__ xg_out)
{
    const int b   = blockIdx.x;
    const int tid = threadIdx.x;
    __shared__ float red[512];
    __shared__ float dh[8];
    __shared__ float alpha[8][1025];   // [h][e], e==1024 is the self loop

    for (int h = 0; h < 8; ++h) {
        const float ad = a_d[b * 8 + h];

        float pm = -3.0e38f;
        for (int e = tid; e < 1024; e += 512) {
            const int i = b * 1024 + e;
            if (i == b) continue;                       // masked self-edge
            float v = a_s[i * 8 + h] + ad;
            v = v > 0.f ? v : 0.2f * v;                 // leaky_relu
            pm = fmaxf(pm, v);
        }
        if (tid == 0) {
            float v = a_s[b * 8 + h] + ad;              // self loop
            v = v > 0.f ? v : 0.2f * v;
            pm = fmaxf(pm, v);
        }
        red[tid] = pm; __syncthreads();
        for (int off = 256; off > 0; off >>= 1) {
            if (tid < off) red[tid] = fmaxf(red[tid], red[tid + off]);
            __syncthreads();
        }
        const float mH = red[0];
        __syncthreads();

        float ps = 0.f;
        for (int e = tid; e < 1024; e += 512) {
            const int i = b * 1024 + e;
            float w = 0.f;
            if (i != b) {
                float v = a_s[i * 8 + h] + ad;
                v = v > 0.f ? v : 0.2f * v;
                w = expf(v - mH);
            }
            alpha[h][e] = w;
            ps += w;
        }
        if (tid == 0) {
            float v = a_s[b * 8 + h] + ad;
            v = v > 0.f ? v : 0.2f * v;
            const float w = expf(v - mH);
            alpha[h][1024] = w;
            ps += w;
        }
        red[tid] = ps; __syncthreads();
        for (int off = 256; off > 0; off >>= 1) {
            if (tid < off) red[tid] += red[tid + off];
            __syncthreads();
        }
        if (tid == 0) dh[h] = red[0];
        __syncthreads();
    }

    const int h = tid >> 6, c = tid & 63;
    const float inv = 1.0f / dh[h];
    float agg = alpha[h][1024] * xs[(size_t)b * 512 + h * 64 + c];   // self loop
    for (int e = 0; e < 1024; ++e)
        agg += alpha[h][e] * xs[((size_t)(b * 1024 + e)) * 512 + h * 64 + c];
    agg *= inv;
    const float v = agg + gat_bias[h * 64 + c] + xg_ln[b * 512 + h * 64 + c];
    xg_out[b * 512 + h * 64 + c] = f2bfu(v);
}

// ---------------------------------------------------------------------------
extern "C" void kernel_launch(void* const* d_in, const int* in_sizes, int n_in,
                              void* d_out, int out_size, void* d_ws, size_t ws_size,
                              hipStream_t stream)
{
    (void)in_sizes; (void)n_in; (void)out_size; (void)ws_size;

    constexpr int NF = 65536, B = 64, IN = 1024, HID = 1024, D = 512, LLM = 2048;

    const float* x_fine   = (const float*)d_in[0];
    const float* x_global = (const float*)d_in[1];
    // d_in[2] = edge_index (star topology is structural; unused)
    const float* nf_g     = (const float*)d_in[3];
    const float* nf_b     = (const float*)d_in[4];
    const float* ip_w1    = (const float*)d_in[5];
    const float* ip_b1    = (const float*)d_in[6];
    const float* ip_w2    = (const float*)d_in[7];
    const float* ip_b2    = (const float*)d_in[8];
    const float* ng_g     = (const float*)d_in[9];
    const float* ng_b     = (const float*)d_in[10];
    const float* gat_ws   = (const float*)d_in[11];
    const float* gat_wd   = (const float*)d_in[12];
    const float* att_s    = (const float*)d_in[13];
    const float* att_d    = (const float*)d_in[14];
    const float* gat_bias = (const float*)d_in[15];
    const float* gp_w     = (const float*)d_in[16];
    const float* gp_b     = (const float*)d_in[17];
    const float* fp_w     = (const float*)d_in[18];
    const float* fp_b     = (const float*)d_in[19];
    float* out = (float*)d_out;

    // ---- workspace layout (~478 MB peak) ----
    char* ws = (char*)d_ws;
    auto take = [&](size_t bytes) { char* p = ws; ws += (bytes + 255) & ~(size_t)255; return p; };
    ushort_t* xf_bf   = (ushort_t*)take((size_t)NF * IN  * 2);   // ln(x_fine) bf16
    ushort_t* h_bf    = (ushort_t*)take((size_t)NF * HID * 2);   // relu MLP hidden bf16
    ushort_t* xf2_bf  = (ushort_t*)take((size_t)NF * D   * 2);   // fine projection bf16
    float*    xs_f    = (float*)   take((size_t)NF * D   * 4);   // xs fp32
    unsigned* w1p     = (unsigned*)take((size_t)IN  / 2 * HID * 4);
    unsigned* w2p     = (unsigned*)take((size_t)HID / 2 * D   * 4);
    unsigned* wsp     = (unsigned*)take((size_t)D   / 2 * D   * 4);
    unsigned* wdp     = (unsigned*)take((size_t)D   / 2 * D   * 4);
    unsigned* gpp     = (unsigned*)take((size_t)D   / 2 * LLM * 4);
    unsigned* fpp     = (unsigned*)take((size_t)D   / 2 * LLM * 4);
    float*    a_sv    = (float*)   take((size_t)NF * 8 * 4);
    float*    xg_ln   = (float*)   take((size_t)B * D * 4);
    ushort_t* xg_lnb  = (ushort_t*)take((size_t)B * D * 2);
    float*    xd      = (float*)   take((size_t)B * D * 4);
    float*    a_dv    = (float*)   take((size_t)B * 8 * 4);
    ushort_t* xg_outb = (ushort_t*)take((size_t)B * D * 2);

    dim3 blk(256);
    auto pgrid = [](long long e) { return dim3((unsigned)((e + 255) / 256)); };

    // 0) pack all weight matrices into WMMA B-fragment layout (one-time, ~8 MB)
    pack_weight<<<pgrid((long long)IN / 2 * HID), blk, 0, stream>>>(ip_w1,  w1p, IN,  HID);
    pack_weight<<<pgrid((long long)HID / 2 * D ), blk, 0, stream>>>(ip_w2,  w2p, HID, D);
    pack_weight<<<pgrid((long long)D / 2 * D   ), blk, 0, stream>>>(gat_ws, wsp, D,   D);
    pack_weight<<<pgrid((long long)D / 2 * D   ), blk, 0, stream>>>(gat_wd, wdp, D,   D);
    pack_weight<<<pgrid((long long)D / 2 * LLM ), blk, 0, stream>>>(gp_w,   gpp, D,   LLM);
    pack_weight<<<pgrid((long long)D / 2 * LLM ), blk, 0, stream>>>(fp_w,   fpp, D,   LLM);

    // 1) LayerNorm fine (bf16 only) & global (fp32 + bf16)
    ln_kernel<<<NF, blk, 0, stream>>>(x_fine,   nf_g, nf_b, nullptr, xf_bf,  IN);
    ln_kernel<<<B,  blk, 0, stream>>>(x_global, ng_g, ng_b, xg_ln,   xg_lnb, D);

    // 2) h = relu(xf_ln @ ip_w1 + b1)            [NF, HID] bf16
    gemm_wmma<1, 0, 0, 1><<<dim3(HID / 64, NF / 128), blk, 0, stream>>>(
        xf_bf, w1p, ip_b1, nullptr, h_bf, NF, HID, IN, 0);

    // 3) xf2 = h @ ip_w2 + b2                    [NF, D] bf16
    gemm_wmma<0, 0, 0, 1><<<dim3(D / 64, NF / 128), blk, 0, stream>>>(
        h_bf, w2p, ip_b2, nullptr, xf2_bf, NF, D, HID, 0);

    // 4) xs = xf2 @ gat_ws                       [NF, D] fp32
    gemm_wmma<0, 0, 1, 0><<<dim3(D / 64, NF / 128), blk, 0, stream>>>(
        xf2_bf, wsp, nullptr, xs_f, nullptr, NF, D, D, D);

    // 5) a_s[i,h]
    attn_dot<<<(NF * 8) / 256, blk, 0, stream>>>(xs_f, att_s, a_sv, NF);

    // 6) xd = xg_ln @ gat_wd ; a_d[b,h]
    gemm_wmma<0, 0, 1, 0><<<dim3(D / 64, 1), blk, 0, stream>>>(
        xg_lnb, wdp, nullptr, xd, nullptr, B, D, D, D);
    attn_dot<<<(B * 8 + 255) / 256, blk, 0, stream>>>(xd, att_d, a_dv, B);

    // 7) segment softmax + aggregation + bias + residual -> xg_out (bf16)
    gat_aggregate<<<B, dim3(512), 0, stream>>>(a_sv, a_dv, xs_f, xg_ln, gat_bias, xg_outb);

    // 8) og = xg_out @ gp_w + gp_b -> out[b, 0, :]
    gemm_wmma<0, 2, 1, 0><<<dim3(LLM / 64, 1), blk, 0, stream>>>(
        xg_outb, gpp, gp_b, out, nullptr, B, LLM, D, LLM);

    // 9) of = xf2 @ fp_w + fp_b -> out[b, 1+p, :]
    gemm_wmma<0, 1, 1, 0><<<dim3(LLM / 64, NF / 128), blk, 0, stream>>>(
        xf2_bf, fpp, fp_b, out, nullptr, NF, LLM, D, LLM);
}
